// CausalGatedSSMBlock_13675175870819
// MI455X (gfx1250) — matrix operand order repaired and checked by reading, not compiled, not run on hardware
//
#include <hip/hip_runtime.h>

// ---------------------------------------------------------------- types
typedef __attribute__((ext_vector_type(8)))  __bf16 v8bf;
typedef __attribute__((ext_vector_type(16))) __bf16 v16bf;
typedef __attribute__((ext_vector_type(8)))  float  v8f;

#define D_MODEL 1024
#define INNER   2048
#define BATCH   4
#define SEQ     2048
#define MROWS   (BATCH * SEQ)   // 8192

__device__ __forceinline__ unsigned short f2bf(float f) {
  unsigned u = __float_as_uint(f);
  u += 0x7fffu + ((u >> 16) & 1u);   // round-to-nearest-even
  return (unsigned short)(u >> 16);
}
__device__ __forceinline__ float sigmoidf_(float x) { return 1.f / (1.f + __expf(-x)); }

// ---------------- CDNA5 async global->LDS copy support (ASYNCcnt path) ----
// builtin signature (from clang diagnostic): param0 = int4 ptr in AS1 (__device__),
// param1 = int4 ptr in AS3 (__shared__), then imm offset, imm cpol.
#if defined(__AMDGCN__) && __has_builtin(__builtin_amdgcn_global_load_async_to_lds_b128)
#define HAVE_ASYNC_LDS 1
typedef int v4i_g __attribute__((vector_size(16)));   // GCC-style int4 (matches builtin)
#define AS1 __attribute__((address_space(1)))
#define AS3 __attribute__((address_space(3)))
__device__ __forceinline__ AS3 v4i_g* lds_cast(void* p) {
  // flat shared pointer: low 32 bits carry the LDS offset
  return (AS3 v4i_g*)(unsigned)(unsigned long long)p;
}
__device__ __forceinline__ AS1 v4i_g* gbl_cast(const void* p) {
  return (AS1 v4i_g*)(unsigned long long)p;
}
#endif

template <int CNT>
__device__ __forceinline__ void async_wait() {
#if defined(HAVE_ASYNC_LDS)
#if __has_builtin(__builtin_amdgcn_s_wait_asynccnt)
  __builtin_amdgcn_s_wait_asynccnt((unsigned short)CNT);
#else
  if (CNT == 0) asm volatile("s_wait_asynccnt 0x0" ::: "memory");
  else          asm volatile("s_wait_asynccnt 0x4" ::: "memory");
#endif
#endif
}

// ------------------------------------------------ weight convert + transpose
// W: f32 [K][N] row-major  ->  Wt: bf16 [N][K]
__global__ __launch_bounds__(256) void wcvt_t_kernel(const float* __restrict__ W,
                                                     unsigned short* __restrict__ Wt,
                                                     int K, int N) {
  size_t i = (size_t)blockIdx.x * 256 + threadIdx.x;   // linear over input [K][N]
  if (i >= (size_t)K * N) return;
  int k = (int)(i / N);
  int n = (int)(i % N);
  Wt[(size_t)n * K + k] = f2bf(W[i]);
}

// ------------------------------------------------------------- layernorm
__global__ __launch_bounds__(256) void layernorm_kernel(const float* __restrict__ x,
                                                        const float* __restrict__ gamma,
                                                        const float* __restrict__ beta,
                                                        unsigned short* __restrict__ xn) {
  __shared__ float red0[8], red1[8];
  const int row = blockIdx.x;
  const float* xr = x + (size_t)row * D_MODEL;
  float s = 0.f, s2 = 0.f;
  for (int i = threadIdx.x; i < D_MODEL; i += 256) { float v = xr[i]; s += v; s2 += v * v; }
  for (int off = 16; off > 0; off >>= 1) { s += __shfl_down(s, off, 32); s2 += __shfl_down(s2, off, 32); }
  const int lane = threadIdx.x & 31, w = threadIdx.x >> 5;
  if (lane == 0) { red0[w] = s; red1[w] = s2; }
  __syncthreads();
  if (w == 0) {
    float a = (lane < 8) ? red0[lane] : 0.f;
    float b = (lane < 8) ? red1[lane] : 0.f;
    for (int off = 4; off > 0; off >>= 1) { a += __shfl_down(a, off, 32); b += __shfl_down(b, off, 32); }
    if (lane == 0) { red0[0] = a; red1[0] = b; }
  }
  __syncthreads();
  const float mu   = red0[0] * (1.f / D_MODEL);
  const float var  = red1[0] * (1.f / D_MODEL) - mu * mu;
  const float rstd = rsqrtf(var + 1e-5f);
  for (int i = threadIdx.x; i < D_MODEL; i += 256) {
    float v = (xr[i] - mu) * rstd * gamma[i] + beta[i];
    xn[(size_t)row * D_MODEL + i] = f2bf(v);
  }
}

// --------------------------------------------------------- WMMA bf16 GEMM
// C[MROWS][N] (f32) = epilogue( A[MROWS][K](bf16) * Bt[N][K](bf16)^T + bias[N] )
// 128x128 block tile, 8 waves, wave tile 64x32 (4x2 wmma accs),
// double-buffered LDS fed by async global->LDS copies.
#define TM 128
#define TN 128
#define TK 32
#define LDT 40   // padded K-stride in elements (80B: 16B aligned, conflict-free banks)

enum { EPI_BIAS = 0, EPI_SIGMOID = 1, EPI_TANH = 2, EPI_RESID = 3 };

template <int EPI, int N, int K>
__global__ __launch_bounds__(256)
void gemm_bf16_wmma(const unsigned short* __restrict__ A,
                    const unsigned short* __restrict__ Bt,
                    const float* __restrict__ bias,
                    const float* __restrict__ resid,
                    float* __restrict__ C) {
  __shared__ unsigned short sA[2][TM * LDT];
  __shared__ unsigned short sB[2][TN * LDT];

  const int t    = threadIdx.x;
  const int lane = t & 31;
  const int wave = t >> 5;
  const int half = lane >> 4;   // 0/1 per WMMA reg layout
  const int l16  = lane & 15;

  const int m_base = blockIdx.y * TM;
  const int n_base = blockIdx.x * TN;
  const int wm = (wave & 1) * 64;   // 2 waves cover 128 rows
  const int wn = (wave >> 1) * 32;  // 4 waves cover 128 cols

  // per-thread tile-load coordinates: 128 rows x 32 bf16, 16B chunks
  const int c0row = t >> 2;               // rows 0..63
  const int c0off = (t & 3) * 8;
  const int c1row = (t + 256) >> 2;       // rows 64..127
  const int c1off = c0off;

  auto load_tiles = [&](int kt, int buf) {
    const unsigned short* ga0 = A  + (m_base + c0row) * K + kt * TK + c0off;
    const unsigned short* gb0 = Bt + (n_base + c0row) * K + kt * TK + c0off;
    const unsigned short* ga1 = A  + (m_base + c1row) * K + kt * TK + c1off;
    const unsigned short* gb1 = Bt + (n_base + c1row) * K + kt * TK + c1off;
    unsigned short* la0 = &sA[buf][c0row * LDT + c0off];
    unsigned short* lb0 = &sB[buf][c0row * LDT + c0off];
    unsigned short* la1 = &sA[buf][c1row * LDT + c1off];
    unsigned short* lb1 = &sB[buf][c1row * LDT + c1off];
#if defined(HAVE_ASYNC_LDS)
    __builtin_amdgcn_global_load_async_to_lds_b128(gbl_cast(ga0), lds_cast(la0), 0, 0);
    __builtin_amdgcn_global_load_async_to_lds_b128(gbl_cast(gb0), lds_cast(lb0), 0, 0);
    __builtin_amdgcn_global_load_async_to_lds_b128(gbl_cast(ga1), lds_cast(la1), 0, 0);
    __builtin_amdgcn_global_load_async_to_lds_b128(gbl_cast(gb1), lds_cast(lb1), 0, 0);
#else
    *(uint4*)la0 = *(const uint4*)ga0;
    *(uint4*)lb0 = *(const uint4*)gb0;
    *(uint4*)la1 = *(const uint4*)ga1;
    *(uint4*)lb1 = *(const uint4*)gb1;
#endif
  };

  v8f acc[4][2];
#pragma unroll
  for (int i = 0; i < 4; ++i)
#pragma unroll
    for (int j = 0; j < 2; ++j)
#pragma unroll
      for (int e = 0; e < 8; ++e) acc[i][j][e] = 0.f;

  const int ksteps = K / TK;
  load_tiles(0, 0);                       // prologue: tile 0 in flight

  for (int kt = 0; kt < ksteps; ++kt) {
    const int cur = kt & 1;
    if (kt + 1 < ksteps) {
      __syncthreads();                    // everyone done reading buf cur^1
      load_tiles(kt + 1, cur ^ 1);        // next tile in flight (async)
      async_wait<4>();                    // my 4 current-tile copies landed
    } else {
      async_wait<0>();
    }
    __syncthreads();                      // all waves' current tile visible

    v16bf af[4], bfr[2];
#pragma unroll
    for (int i = 0; i < 4; ++i) {
      // A 16x32 layout: M=l16, K = 16*half .. 16*half+15
      const unsigned short* p = &sA[cur][(wm + 16 * i + l16) * LDT];
      v8bf lo = *(const v8bf*)(p + 16 * half);
      v8bf hi = *(const v8bf*)(p + 16 * half + 8);
      af[i] = __builtin_shufflevector(lo, hi, 0, 1, 2, 3, 4, 5, 6, 7, 8, 9, 10, 11, 12, 13, 14, 15);
    }
#pragma unroll
    for (int j = 0; j < 2; ++j) {
      // B 32x16 layout: N=l16, K = 16*half .. 16*half+15
      const unsigned short* p = &sB[cur][(wn + 16 * j + l16) * LDT + 16 * half];
      v8bf lo = *(const v8bf*)(p);
      v8bf hi = *(const v8bf*)(p + 8);
      bfr[j] = __builtin_shufflevector(lo, hi, 0, 1, 2, 3, 4, 5, 6, 7, 8, 9, 10, 11, 12, 13, 14, 15);
    }
#pragma unroll
    for (int i = 0; i < 4; ++i)
#pragma unroll
      for (int j = 0; j < 2; ++j)
        acc[i][j] = __builtin_amdgcn_wmma_f32_16x16x32_bf16(
            false, af[i], false, bfr[j], (short)0, acc[i][j], false, false);
  }

  // epilogue: C/D layout -> VGPR r: M = 8*half + r (+subtile), N = l16 (+subtile)
#pragma unroll
  for (int i = 0; i < 4; ++i) {
#pragma unroll
    for (int j = 0; j < 2; ++j) {
      const int col = n_base + wn + 16 * j + l16;
      const float bv = bias[col];
#pragma unroll
      for (int r = 0; r < 8; ++r) {
        const int row = m_base + wm + 16 * i + 8 * half + r;
        float v = acc[i][j][r] + bv;
        if (EPI == EPI_SIGMOID)    v = sigmoidf_(v);
        else if (EPI == EPI_TANH)  v = tanhf(v);
        else if (EPI == EPI_RESID) v += resid[row * N + col];
        C[row * N + col] = v;
      }
    }
  }
}

// ------------------------------------- causal depthwise conv (K=3) + gating
__global__ __launch_bounds__(256) void conv_gate_kernel(const float* __restrict__ proj,
                                                        const float* __restrict__ conv_w,
                                                        const float* __restrict__ conv_b,
                                                        float* __restrict__ uf,
                                                        unsigned short* __restrict__ ub) {
  size_t idx = (size_t)blockIdx.x * 256 + threadIdx.x;  // over MROWS*INNER
  if (idx >= (size_t)MROWS * INNER) return;
  const int ch  = (int)(idx % INNER);
  const int s   = (int)((idx / INNER) % SEQ);
  const int b   = (int)(idx / ((size_t)INNER * SEQ));
  const size_t row = (size_t)b * SEQ + s;

  const float w0 = conv_w[ch * 3 + 0];
  const float w1 = conv_w[ch * 3 + 1];
  const float w2 = conv_w[ch * 3 + 2];
  const size_t stride = 2 * INNER;
  const float* p = proj + row * stride + ch;

  float acc = conv_b[ch] + w2 * p[0];
  if (s >= 1) acc += w1 * p[-(ptrdiff_t)stride];
  if (s >= 2) acc += w0 * p[-(ptrdiff_t)(2 * stride)];

  const float gate = proj[row * stride + INNER + ch];
  const float u = (acc * sigmoidf_(acc)) * sigmoidf_(gate);   // silu(conv)*sigmoid(gate)
  uf[row * INNER + ch] = u;
  ub[row * INNER + ch] = f2bf(u);
}

// ----------------------------------------------- linear recurrence over time
__global__ __launch_bounds__(256) void scan_kernel(const float* __restrict__ av,
                                                   const float* __restrict__ bv,
                                                   const float* __restrict__ cv,
                                                   const float* __restrict__ uf,
                                                   unsigned short* __restrict__ yb) {
  const int tid = blockIdx.x * 256 + threadIdx.x;       // 0..BATCH*INNER-1
  const int b  = tid / INNER;
  const int ch = tid % INNER;
  float st = 0.f;
  size_t off = ((size_t)b * SEQ) * INNER + ch;
  for (int s = 0; s < SEQ; ++s, off += INNER) {
    const float u = uf[off];
    st = av[off] * st + bv[off] * u;
    yb[off] = f2bf(cv[off] * st + u);
  }
}

// ---------------------------------------------------------------- launcher
extern "C" void kernel_launch(void* const* d_in, const int* in_sizes, int n_in,
                              void* d_out, int out_size, void* d_ws, size_t ws_size,
                              hipStream_t stream) {
  const float* x      = (const float*)d_in[0];
  const float* W_in   = (const float*)d_in[1];
  const float* b_in   = (const float*)d_in[2];
  const float* conv_w = (const float*)d_in[3];
  const float* conv_b = (const float*)d_in[4];
  const float* Wa     = (const float*)d_in[5];
  const float* ba     = (const float*)d_in[6];
  const float* Wb     = (const float*)d_in[7];
  const float* bb     = (const float*)d_in[8];
  const float* Wc     = (const float*)d_in[9];
  const float* bc     = (const float*)d_in[10];
  const float* Wo     = (const float*)d_in[11];
  const float* bo     = (const float*)d_in[12];
  const float* gamma  = (const float*)d_in[13];
  const float* beta   = (const float*)d_in[14];
  float* out = (float*)d_out;

  // ---- workspace carve-up (256B aligned)
  char* ws = (char*)d_ws;
  size_t o = 0;
  auto carve = [&](size_t bytes) { size_t r = o; o += (bytes + 255) & ~(size_t)255; return r; };
  unsigned short* xn   = (unsigned short*)(ws + carve((size_t)MROWS * D_MODEL * 2));
  unsigned short* WtIn = (unsigned short*)(ws + carve((size_t)(2 * INNER) * D_MODEL * 2));
  unsigned short* WtA  = (unsigned short*)(ws + carve((size_t)INNER * INNER * 2));
  unsigned short* WtB  = (unsigned short*)(ws + carve((size_t)INNER * INNER * 2));
  unsigned short* WtC  = (unsigned short*)(ws + carve((size_t)INNER * INNER * 2));
  unsigned short* WtO  = (unsigned short*)(ws + carve((size_t)D_MODEL * INNER * 2));
  float*          proj = (float*)(ws + carve((size_t)MROWS * 2 * INNER * 4));   // reused for a,b
  float*          uf   = (float*)(ws + carve((size_t)MROWS * INNER * 4));
  unsigned short* ub   = (unsigned short*)(ws + carve((size_t)MROWS * INNER * 2));
  float*          cbuf = (float*)(ws + carve((size_t)MROWS * INNER * 4));
  unsigned short* yb   = (unsigned short*)(ws + carve((size_t)MROWS * INNER * 2));
  float* abuf = proj;                                  // proj dead after conv
  float* bbuf = proj + (size_t)MROWS * INNER;

  // ---- 1. weight convert+transpose (f32 [K][N] -> bf16 [N][K])
  {
    size_t n;
    n = (size_t)D_MODEL * 2 * INNER;
    wcvt_t_kernel<<<(unsigned)((n + 255) / 256), 256, 0, stream>>>(W_in, WtIn, D_MODEL, 2 * INNER);
    n = (size_t)INNER * INNER;
    wcvt_t_kernel<<<(unsigned)((n + 255) / 256), 256, 0, stream>>>(Wa, WtA, INNER, INNER);
    wcvt_t_kernel<<<(unsigned)((n + 255) / 256), 256, 0, stream>>>(Wb, WtB, INNER, INNER);
    wcvt_t_kernel<<<(unsigned)((n + 255) / 256), 256, 0, stream>>>(Wc, WtC, INNER, INNER);
    n = (size_t)INNER * D_MODEL;
    wcvt_t_kernel<<<(unsigned)((n + 255) / 256), 256, 0, stream>>>(Wo, WtO, INNER, D_MODEL);
  }

  // ---- 2. layernorm -> bf16 xn
  layernorm_kernel<<<MROWS, 256, 0, stream>>>(x, gamma, beta, xn);

  // ---- 3. proj = xn @ W_in + b_in        [8192 x 4096], K=1024
  gemm_bf16_wmma<EPI_BIAS, 2 * INNER, D_MODEL>
      <<<dim3((2 * INNER) / TN, MROWS / TM), 256, 0, stream>>>(xn, WtIn, b_in, nullptr, proj);

  // ---- 4. causal conv + gate -> u (f32 + bf16)
  conv_gate_kernel<<<(unsigned)(((size_t)MROWS * INNER + 255) / 256), 256, 0, stream>>>(
      proj, conv_w, conv_b, uf, ub);

  // ---- 5. a,b,c gates                    [8192 x 2048], K=2048
  gemm_bf16_wmma<EPI_SIGMOID, INNER, INNER>
      <<<dim3(INNER / TN, MROWS / TM), 256, 0, stream>>>(ub, WtA, ba, nullptr, abuf);
  gemm_bf16_wmma<EPI_SIGMOID, INNER, INNER>
      <<<dim3(INNER / TN, MROWS / TM), 256, 0, stream>>>(ub, WtB, bb, nullptr, bbuf);
  gemm_bf16_wmma<EPI_TANH, INNER, INNER>
      <<<dim3(INNER / TN, MROWS / TM), 256, 0, stream>>>(ub, WtC, bc, nullptr, cbuf);

  // ---- 6. sequential scan over time
  scan_kernel<<<(BATCH * INNER) / 256, 256, 0, stream>>>(abuf, bbuf, cbuf, uf, yb);

  // ---- 7. out = x + y @ Wo + bo          [8192 x 1024], K=2048
  gemm_bf16_wmma<EPI_RESID, D_MODEL, INNER>
      <<<dim3(D_MODEL / TN, MROWS / TM), 256, 0, stream>>>(yb, WtO, bo, x, out);
}
